// TGNPLInventory_74801150427802
// MI455X (gfx1250) — matrix operand order repaired and checked by reading, not compiled, not run on hardware
//
#include <hip/hip_runtime.h>
#include <hip/hip_bf16.h>

typedef __attribute__((ext_vector_type(16))) _Float16 v16h;
typedef __attribute__((ext_vector_type(8)))  float    v8f;

#define P_DIM   500
#define P_PAD   512
#define D_DIM   128
#define DEBT_PEN 10.0

// ---------------------------------------------------------------------------
// 1) zero TS (padded F x 512 f32) + the two f64 accumulators
// ---------------------------------------------------------------------------
__global__ void zero_kernel(float4* __restrict__ ts4, long n4, double* __restrict__ accum) {
    long i = (long)blockIdx.x * blockDim.x + threadIdx.x;
    long stride = (long)gridDim.x * blockDim.x;
    float4 z = make_float4(0.f, 0.f, 0.f, 0.f);
    for (; i < n4; i += stride) ts4[i] = z;
    if (blockIdx.x == 0 && threadIdx.x == 0) { accum[0] = 0.0; accum[1] = 0.0; }
}

// ---------------------------------------------------------------------------
// 2) scatter-add amounts into TS[src, prod-F]
// ---------------------------------------------------------------------------
__global__ void scatter_kernel(const int* __restrict__ src, const int* __restrict__ prod,
                               const float* __restrict__ raw, float* __restrict__ TS,
                               int nE, int nF) {
    int e = blockIdx.x * blockDim.x + threadIdx.x;
    if (e >= nE) return;
    float amt = fmaxf(raw[(size_t)e * 4], 1.0f);
    int f = src[e];
    int p = prod[e] - nF;
    atomicAdd(&TS[(size_t)f * P_PAD + p], amt);
}

// ---------------------------------------------------------------------------
// 3) M1[k,d] = sum_j E[k,j] * B[j,d]      (500 x 128)
// ---------------------------------------------------------------------------
__global__ void m1_kernel(const float* __restrict__ E, const float* __restrict__ B,
                          float* __restrict__ M1) {
    int idx = blockIdx.x * blockDim.x + threadIdx.x;
    if (idx >= P_DIM * D_DIM) return;
    int k = idx >> 7, d = idx & 127;
    float s = 0.f;
    #pragma unroll 8
    for (int j = 0; j < D_DIM; ++j) s += E[k * D_DIM + j] * B[j * D_DIM + d];
    M1[idx] = s;
}

// ---------------------------------------------------------------------------
// 4) Wt[n,k] = relu( dot(M1[k,:], E[n,:]) )  (== W[k,n]),  padded 512x512 f16
// ---------------------------------------------------------------------------
__global__ void wt_kernel(const float* __restrict__ M1, const float* __restrict__ E,
                          _Float16* __restrict__ Wt) {
    int idx = blockIdx.x * blockDim.x + threadIdx.x;
    if (idx >= P_PAD * P_PAD) return;
    int n = idx >> 9, k = idx & 511;
    float s = 0.f;
    if (n < P_DIM && k < P_DIM) {
        #pragma unroll 8
        for (int d = 0; d < D_DIM; ++d) s += M1[k * D_DIM + d] * E[n * D_DIM + d];
        s = fmaxf(s, 0.f);
    }
    Wt[idx] = (_Float16)s;
}

// ---------------------------------------------------------------------------
// 5) Fused GEMM + reduction.
//    Block = 256 threads = 8 waves. Block b owns rows [16b, 16b+16).
//    Wave w covers N-columns [64w, 64w+64) as 4 WMMA tiles.
//    K loop: 16 steps of K=32 using v_wmma_f32_16x16x32_f16.
//    Epilogue: debt/cons partial sums -> LDS reduce -> f64 atomics.
// ---------------------------------------------------------------------------
__global__ __launch_bounds__(256)
void gemm_reduce_kernel(const float* __restrict__ TS, const _Float16* __restrict__ Wt,
                        const float* __restrict__ inv, double* __restrict__ accum) {
    const int m0    = blockIdx.x * 16;
    const int lane  = threadIdx.x & 31;
    const int wave  = threadIdx.x >> 5;
    const int rowin = lane & 15;
    const int hsel  = lane >> 4;           // 0 or 1
    const int n0    = wave * 64;

    // A: lane holds row (m0+rowin); element i -> K = hsel*16 + i
    const float* tsrow = TS + (size_t)(m0 + rowin) * P_PAD + hsel * 16;
    // B: lane holds column n; element i -> K = hsel*16 + i  (contiguous in Wt row)
    const _Float16* w0 = Wt + (size_t)(n0 +  0 + rowin) * P_PAD + hsel * 16;
    const _Float16* w1 = w0 + 16 * P_PAD;
    const _Float16* w2 = w1 + 16 * P_PAD;
    const _Float16* w3 = w2 + 16 * P_PAD;

    v8f c0 = {}, c1 = {}, c2 = {}, c3 = {};

    for (int kt = 0; kt < P_PAD / 32; ++kt) {
        const int kb = kt * 32;
        const float4* p = (const float4*)(tsrow + kb);
        float4 x0 = p[0], x1 = p[1];
        const float4* q = (const float4*)(tsrow + kb + 8);
        float4 x2 = q[0], x3 = q[1];

        v16h a;
        a[0]  = (_Float16)x0.x; a[1]  = (_Float16)x0.y;
        a[2]  = (_Float16)x0.z; a[3]  = (_Float16)x0.w;
        a[4]  = (_Float16)x1.x; a[5]  = (_Float16)x1.y;
        a[6]  = (_Float16)x1.z; a[7]  = (_Float16)x1.w;
        a[8]  = (_Float16)x2.x; a[9]  = (_Float16)x2.y;
        a[10] = (_Float16)x2.z; a[11] = (_Float16)x2.w;
        a[12] = (_Float16)x3.x; a[13] = (_Float16)x3.y;
        a[14] = (_Float16)x3.z; a[15] = (_Float16)x3.w;

        v16h b0 = *(const v16h*)(w0 + kb);
        v16h b1 = *(const v16h*)(w1 + kb);
        v16h b2 = *(const v16h*)(w2 + kb);
        v16h b3 = *(const v16h*)(w3 + kb);

        c0 = __builtin_amdgcn_wmma_f32_16x16x32_f16(false, a, false, b0, (short)0, c0, false, false);
        c1 = __builtin_amdgcn_wmma_f32_16x16x32_f16(false, a, false, b1, (short)0, c1, false, false);
        c2 = __builtin_amdgcn_wmma_f32_16x16x32_f16(false, a, false, b2, (short)0, c2, false, false);
        c3 = __builtin_amdgcn_wmma_f32_16x16x32_f16(false, a, false, b3, (short)0, c3, false, false);
    }

    // Epilogue: C/D layout -> row = m0 + v + hsel*8, col = n0 + j*16 + (lane&15)
    float debt = 0.f, cons = 0.f;
    const int col_lane = lane & 15;
    const int mrow = m0 + hsel * 8;

#define EPILOGUE(C, J)                                                          \
    do {                                                                        \
        int col = n0 + (J) * 16 + col_lane;                                     \
        if (col < P_DIM) {                                                      \
            _Pragma("unroll")                                                   \
            for (int v = 0; v < 8; ++v) {                                       \
                float tc = (C)[v];                                              \
                cons += tc;                                                     \
                float iv = inv[(size_t)(mrow + v) * P_DIM + col];               \
                debt += fmaxf(tc - iv, 0.f);                                    \
            }                                                                   \
        }                                                                       \
    } while (0)

    EPILOGUE(c0, 0); EPILOGUE(c1, 1); EPILOGUE(c2, 2); EPILOGUE(c3, 3);
#undef EPILOGUE

    __shared__ float sdebt[256];
    __shared__ float scons[256];
    const int tid = threadIdx.x;
    sdebt[tid] = debt; scons[tid] = cons;
    __syncthreads();
    for (int s = 128; s > 0; s >>= 1) {
        if (tid < s) { sdebt[tid] += sdebt[tid + s]; scons[tid] += scons[tid + s]; }
        __syncthreads();
    }
    if (tid == 0) {
        atomicAdd(&accum[0], (double)sdebt[0]);   // raw debt sum (pre x10)
        atomicAdd(&accum[1], (double)scons[0]);   // consumption sum
    }
}

// ---------------------------------------------------------------------------
// 6) finalize: three scalars
// ---------------------------------------------------------------------------
__global__ void finalize_kernel(const double* __restrict__ accum, float* __restrict__ out,
                                double inv_n) {
    if (blockIdx.x == 0 && threadIdx.x == 0) {
        double debt10 = DEBT_PEN * accum[0];
        double cons   = accum[1];
        out[0] = (float)((debt10 - cons) * inv_n);
        out[1] = (float)(debt10 * inv_n);
        out[2] = (float)(cons * inv_n);
    }
}

extern "C" void kernel_launch(void* const* d_in, const int* in_sizes, int n_in,
                              void* d_out, int out_size, void* d_ws, size_t ws_size,
                              hipStream_t stream) {
    const int*   src  = (const int*)d_in[0];
    // d_in[1] = dst (unused by the reference)
    const int*   prod = (const int*)d_in[2];
    const float* raw  = (const float*)d_in[3];   // [E,4]
    const float* Eemb = (const float*)d_in[4];   // [500,128]
    const float* Bmat = (const float*)d_in[5];   // [128,128]
    const float* inv  = (const float*)d_in[6];   // [F,500]
    float* out = (float*)d_out;

    const int nE = in_sizes[0];
    const int nF = in_sizes[6] / P_DIM;          // 100000

    // workspace layout
    char* ws = (char*)d_ws;
    size_t off = 0;
    float* TS = (float*)(ws + off);              off += (size_t)nF * P_PAD * sizeof(float);
    off = (off + 255) & ~(size_t)255;
    _Float16* Wt = (_Float16*)(ws + off);        off += (size_t)P_PAD * P_PAD * sizeof(_Float16);
    off = (off + 255) & ~(size_t)255;
    float* M1 = (float*)(ws + off);              off += (size_t)P_DIM * D_DIM * sizeof(float);
    off = (off + 255) & ~(size_t)255;
    double* accum = (double*)(ws + off);

    const long n4 = (long)nF * P_PAD / 4;        // float4 count of TS
    zero_kernel<<<4096, 256, 0, stream>>>((float4*)TS, n4, accum);

    scatter_kernel<<<(nE + 255) / 256, 256, 0, stream>>>(src, prod, raw, TS, nE, nF);

    m1_kernel<<<(P_DIM * D_DIM + 255) / 256, 256, 0, stream>>>(Eemb, Bmat, M1);

    wt_kernel<<<(P_PAD * P_PAD + 255) / 256, 256, 0, stream>>>(M1, Eemb, Wt);

    gemm_reduce_kernel<<<nF / 16, 256, 0, stream>>>(TS, Wt, inv, accum);

    finalize_kernel<<<1, 32, 0, stream>>>(accum, out, 1.0 / (double)nE);
}
